// SpatialTransformerEncoderLayer_21328807592601
// MI455X (gfx1250) — matrix operand order repaired and checked by reading, not compiled, not run on hardware
//
#include <hip/hip_runtime.h>
#include <stdint.h>

#define DD      256
#define HEADS   8
#define LEVELS  4
#define POINTS  4
#define HD      32
#define LQ      12240
#define NB      2
#define NL      (NB * LQ)          // 24480 rows
#define NPROJ   640                // 256 (value) + 256 (offsets) + 128 (attn)

typedef __attribute__((ext_vector_type(16))) __bf16 v16bf;
typedef __attribute__((ext_vector_type(8)))  float  v8f;

union FragAB { v16bf v; uint32_t u[8]; };

__device__ __forceinline__ unsigned short f2bf(float f) {
  union { float f; uint32_t u; } x; x.f = f;
  uint32_t r = x.u + 0x7FFFu + ((x.u >> 16) & 1u);   // round-to-nearest-even
  return (unsigned short)(r >> 16);
}

// ---------------- conversion kernels ----------------
__global__ void conv_src_kernel(const float* __restrict__ src, unsigned short* __restrict__ dst) {
  size_t i = (size_t)blockIdx.x * 256 + threadIdx.x;
  dst[i] = f2bf(src[i]);
}

__global__ void conv_wproj_kernel(const float* __restrict__ wv, const float* __restrict__ wo,
                                  const float* __restrict__ wa, unsigned short* __restrict__ B) {
  int k = blockIdx.x;          // 0..255
  int c = threadIdx.x;         // 0..639
  float v;
  if (c < 256)       v = wv[(size_t)k * 256 + c];
  else if (c < 512)  v = wo[(size_t)k * 256 + (c - 256)];
  else               v = wa[(size_t)k * 128 + (c - 512)];
  B[(size_t)k * NPROJ + c] = f2bf(v);
}

__global__ void conv_wout_kernel(const float* __restrict__ w, unsigned short* __restrict__ B) {
  size_t i = (size_t)blockIdx.x * 256 + threadIdx.x;
  B[i] = f2bf(w[i]);
}

// ---------------- fused projection GEMM: (NL x 256) x (256 x 640) ----------------
// wave-per-16x16-tile; K loop 256/32 = 8 WMMAs; bias fused in epilogue.
__global__ void gemm_proj_kernel(const unsigned short* __restrict__ Abf,   // NL x 256 bf16
                                 const unsigned short* __restrict__ Bbf,   // 256 x 640 bf16
                                 const float* __restrict__ b_value,
                                 const float* __restrict__ b_off,
                                 const float* __restrict__ b_attn,
                                 float* __restrict__ valbuf,               // NL x 256
                                 float* __restrict__ offbuf,               // NL x 256
                                 float* __restrict__ attnbuf) {            // NL x 128
  const int lane = threadIdx.x;            // 0..31
  const int wave = threadIdx.y;            // 0..7
  const int mtile = blockIdx.x * 16;
  const int ntile = (blockIdx.y * 8 + wave) * 16;
  const int K = 256, ldb = NPROJ;
  const int half = lane >> 4;
  const int r = lane & 15;

  const uint32_t* Arow = (const uint32_t*)(Abf + (size_t)(mtile + r) * K);  // 128 dwords/row
  v8f c = {};
  for (int kb = 0; kb < K; kb += 32) {
    FragAB a, b;
#pragma unroll
    for (int j = 0; j < 8; ++j) {
      int k = kb + 8 * half + ((j < 4) ? 2 * j : 16 + 2 * (j - 4));   // ISA A-layout (16-bit 16x32)
      a.u[j] = Arow[k >> 1];
    }
    const unsigned short* Bcol = Bbf + ntile + r;                     // ISA B-layout: lane = N, elements = K (same order as A)
#pragma unroll
    for (int j = 0; j < 8; ++j) {
      int k = kb + 8 * half + ((j < 4) ? 2 * j : 16 + 2 * (j - 4));   // VGPR j = K pair (k,k+1)
      b.u[j] = (uint32_t)Bcol[(size_t)k * ldb] | ((uint32_t)Bcol[(size_t)(k + 1) * ldb] << 16);
    }
    if (kb + 32 < K) __builtin_prefetch(Arow + ((kb + 32) >> 1), 0, 3);
    c = __builtin_amdgcn_wmma_f32_16x16x32_bf16(false, a.v, false, b.v, (short)0, c, false, false);
  }

  // epilogue: route column to the right output buffer, add bias
  int col = ntile + r;
  float bias; float* dst; int dcol, ldc;
  if (col < 256)        { bias = b_value[col];        dst = valbuf;  dcol = col;        ldc = 256; }
  else if (col < 512)   { bias = b_off[col - 256];    dst = offbuf;  dcol = col - 256;  ldc = 256; }
  else                  { bias = b_attn[col - 512];   dst = attnbuf; dcol = col - 512;  ldc = 128; }
#pragma unroll
  for (int v = 0; v < 8; ++v) {
    int m = mtile + v + 8 * half;                                     // ISA D-layout
    dst[(size_t)m * ldc + dcol] = c[v] + bias;
  }
}

// ---------------- softmax over 16 (level,point) logits per (token,head) ----------------
__global__ void softmax16_kernel(float* __restrict__ attn) {
  int idx = blockIdx.x * blockDim.x + threadIdx.x;
  if (idx >= NL * HEADS) return;
  float* p = attn + (size_t)idx * 16;
  float m = p[0];
#pragma unroll
  for (int i = 1; i < 16; ++i) m = fmaxf(m, p[i]);
  float e[16], s = 0.f;
#pragma unroll
  for (int i = 0; i < 16; ++i) { e[i] = __expf(p[i] - m); s += e[i]; }
  float inv = 1.f / s;
#pragma unroll
  for (int i = 0; i < 16; ++i) p[i] = e[i] * inv;
}

// ---------------- deformable bilinear sampling + weighted sum ----------------
// one block per global token; thread = head*32 + channel
__global__ void sample_kernel(const float* __restrict__ value,   // NL x 256 (head-major cols)
                              const float* __restrict__ offs,    // NL x 256
                              const float* __restrict__ attnw,   // NL x 128 (post-softmax)
                              const float* __restrict__ refp,    // NL x 4 x 2
                              unsigned short* __restrict__ outbf) { // NL x 256 bf16
  const int t = blockIdx.x;            // 0..NL-1  (n*LQ + tok)
  const int c = threadIdx.x & 31;
  const int h = threadIdx.x >> 5;
  const int n = t / LQ;
  const int Hs[4] = {96, 48, 24, 12};
  const int Ws[4] = {96, 48, 24, 12};
  const int S0[4] = {0, 9216, 11520, 12096};

  const float* orow = offs  + (size_t)t * 256 + h * 32;
  const float* arow = attnw + (size_t)t * 128 + h * 16;
  const float* rrow = refp  + (size_t)t * 8;

  float acc = 0.f;
#pragma unroll
  for (int l = 0; l < LEVELS; ++l) {
    const int Hi = Hs[l], Wi = Ws[l];
    const float Wf = (float)Wi, Hf = (float)Hi;
    const float rx = rrow[l * 2 + 0], ry = rrow[l * 2 + 1];
    const float* vbase = value + ((size_t)(n * LQ + S0[l])) * 256 + h * 32 + c;
#pragma unroll
    for (int p = 0; p < POINTS; ++p) {
      float lx = rx + orow[l * 8 + p * 2 + 0] / Wf;
      float ly = ry + orow[l * 8 + p * 2 + 1] / Hf;
      float x = lx * Wf - 0.5f;
      float y = ly * Hf - 0.5f;
      float x0f = floorf(x), y0f = floorf(y);
      int   x0 = (int)x0f,  y0 = (int)y0f;
      float fx = x - x0f,   fy = y - y0f;
      int x1 = x0 + 1, y1 = y0 + 1;
      float v00 = 0.f, v10 = 0.f, v01 = 0.f, v11 = 0.f;
      bool xin0 = (x0 >= 0) & (x0 < Wi), xin1 = (x1 >= 0) & (x1 < Wi);
      bool yin0 = (y0 >= 0) & (y0 < Hi), yin1 = (y1 >= 0) & (y1 < Hi);
      if (xin0 & yin0) v00 = vbase[(size_t)(y0 * Wi + x0) * 256];
      if (xin1 & yin0) v10 = vbase[(size_t)(y0 * Wi + x1) * 256];
      if (xin0 & yin1) v01 = vbase[(size_t)(y1 * Wi + x0) * 256];
      if (xin1 & yin1) v11 = vbase[(size_t)(y1 * Wi + x1) * 256];
      float s = (1.f - fx) * (1.f - fy) * v00 + fx * (1.f - fy) * v10
              + (1.f - fx) * fy * v01 + fx * fy * v11;
      acc += arow[l * 4 + p] * s;
    }
  }
  outbf[(size_t)t * 256 + h * 32 + c] = f2bf(acc);
}

// ---------------- output GEMM: (NL x 256) x (256 x 256) + b_out ----------------
__global__ void gemm_out_kernel(const unsigned short* __restrict__ Abf,   // NL x 256 bf16
                                const unsigned short* __restrict__ Bbf,   // 256 x 256 bf16
                                const float* __restrict__ b_out,
                                float* __restrict__ tmp2) {               // NL x 256 f32
  const int lane = threadIdx.x;
  const int wave = threadIdx.y;
  const int mtile = blockIdx.x * 16;
  const int ntile = (blockIdx.y * 8 + wave) * 16;
  const int K = 256, ldb = 256;
  const int half = lane >> 4;
  const int r = lane & 15;

  const uint32_t* Arow = (const uint32_t*)(Abf + (size_t)(mtile + r) * K);
  v8f c = {};
  for (int kb = 0; kb < K; kb += 32) {
    FragAB a, b;
#pragma unroll
    for (int j = 0; j < 8; ++j) {
      int k = kb + 8 * half + ((j < 4) ? 2 * j : 16 + 2 * (j - 4));
      a.u[j] = Arow[k >> 1];
    }
    const unsigned short* Bcol = Bbf + ntile + r;
#pragma unroll
    for (int j = 0; j < 8; ++j) {
      int k = kb + 8 * half + ((j < 4) ? 2 * j : 16 + 2 * (j - 4));
      b.u[j] = (uint32_t)Bcol[(size_t)k * ldb] | ((uint32_t)Bcol[(size_t)(k + 1) * ldb] << 16);
    }
    c = __builtin_amdgcn_wmma_f32_16x16x32_bf16(false, a.v, false, b.v, (short)0, c, false, false);
  }
  int col = ntile + r;
  float bias = b_out[col];
#pragma unroll
  for (int v = 0; v < 8; ++v) {
    int m = mtile + v + 8 * half;
    tmp2[(size_t)m * 256 + col] = c[v] + bias;
  }
}

// ---------------- residual + LayerNorm ----------------
__global__ void ln_kernel(const float* __restrict__ src, const float* __restrict__ tmp2,
                          const float* __restrict__ gamma, const float* __restrict__ beta,
                          float* __restrict__ out) {
  const int row = blockIdx.x;
  const int i = threadIdx.x;            // 256 threads
  __shared__ float sm[256];
  float x = src[(size_t)row * 256 + i] + tmp2[(size_t)row * 256 + i];

  sm[i] = x; __syncthreads();
  for (int s = 128; s > 0; s >>= 1) { if (i < s) sm[i] += sm[i + s]; __syncthreads(); }
  float mu = sm[0] * (1.f / 256.f);
  __syncthreads();

  float d = x - mu;
  sm[i] = d * d; __syncthreads();
  for (int s = 128; s > 0; s >>= 1) { if (i < s) sm[i] += sm[i + s]; __syncthreads(); }
  float var = sm[0] * (1.f / 256.f);

  out[(size_t)row * 256 + i] = d * rsqrtf(var + 1e-5f) * gamma[i] + beta[i];
}

// ---------------- host launcher ----------------
extern "C" void kernel_launch(void* const* d_in, const int* in_sizes, int n_in,
                              void* d_out, int out_size, void* d_ws, size_t ws_size,
                              hipStream_t stream) {
  const float* src     = (const float*)d_in[0];
  const float* refp    = (const float*)d_in[1];
  // d_in[2] spatial_shapes / d_in[3] level_start_index: compile-time constants, hardcoded
  const float* w_value = (const float*)d_in[4];
  const float* b_value = (const float*)d_in[5];
  const float* w_off   = (const float*)d_in[6];
  const float* b_off   = (const float*)d_in[7];
  const float* w_attn  = (const float*)d_in[8];
  const float* b_attn  = (const float*)d_in[9];
  const float* w_out   = (const float*)d_in[10];
  const float* b_out   = (const float*)d_in[11];
  const float* gamma   = (const float*)d_in[12];
  const float* beta    = (const float*)d_in[13];
  float* out = (float*)d_out;

  char* ws = (char*)d_ws;
  size_t pos = 0;
  auto carve = [&](size_t bytes) -> void* {
    void* p = ws + pos;
    pos = (pos + bytes + 255) & ~(size_t)255;
    return p;
  };
  unsigned short* srcbf   = (unsigned short*)carve((size_t)NL * DD * 2);
  unsigned short* wprojbf = (unsigned short*)carve((size_t)DD * NPROJ * 2);
  unsigned short* woutbf  = (unsigned short*)carve((size_t)DD * DD * 2);
  float*          valbuf  = (float*)carve((size_t)NL * DD * 4);
  float*          offbuf  = (float*)carve((size_t)NL * DD * 4);
  float*          attnbuf = (float*)carve((size_t)NL * 128 * 4);
  unsigned short* outbf   = (unsigned short*)carve((size_t)NL * DD * 2);
  float*          tmp2    = (float*)carve((size_t)NL * DD * 4);

  conv_src_kernel  <<<NL, 256, 0, stream>>>(src, srcbf);
  conv_wproj_kernel<<<DD, NPROJ, 0, stream>>>(w_value, w_off, w_attn, wprojbf);
  conv_wout_kernel <<<DD, 256, 0, stream>>>(w_out, woutbf);

  gemm_proj_kernel<<<dim3(NL / 16, NPROJ / 128), dim3(32, 8), 0, stream>>>(
      srcbf, wprojbf, b_value, b_off, b_attn, valbuf, offbuf, attnbuf);

  softmax16_kernel<<<(NL * HEADS + 255) / 256, 256, 0, stream>>>(attnbuf);

  sample_kernel<<<NL, 256, 0, stream>>>(valbuf, offbuf, attnbuf, refp, outbf);

  gemm_out_kernel<<<dim3(NL / 16, DD / 128), dim3(32, 8), 0, stream>>>(
      outbf, woutbf, b_out, tmp2);

  ln_kernel<<<NL, 256, 0, stream>>>(src, tmp2, gamma, beta, out);
}
